// QuantizedChannelMixer_36335423324366
// MI455X (gfx1250) — 8-bit matrix operand order repaired and checked by reading, compiled, not run on hardware
//
#include <hip/hip_runtime.h>
#include <hip/hip_bf16.h>
#include <stdint.h>

typedef int v8i __attribute__((ext_vector_type(8)));
typedef int v4i __attribute__((ext_vector_type(4)));

static constexpr int DMODEL = 2048;
static constexpr int HIDDEN = 8192;
static constexpr int NGV    = 2 * HIDDEN;
static constexpr int TOKENS = 2 * 2048;
static constexpr float EPSF = 1e-5f;
static constexpr float QBF  = 127.0f;

__global__ __launch_bounds__(256) void absum_partial_kernel(
    const float* __restrict__ w, long long n, float* __restrict__ partial) {
  __shared__ float sm[256];
  float s = 0.f;
  long long i = (long long)blockIdx.x * blockDim.x + threadIdx.x;
  const long long stride = (long long)gridDim.x * blockDim.x;
  for (; i < n; i += stride) s += fabsf(w[i]);
  sm[threadIdx.x] = s;
  __syncthreads();
  for (int o = 128; o > 0; o >>= 1) {
    if ((int)threadIdx.x < o) sm[threadIdx.x] += sm[threadIdx.x + o];
    __syncthreads();
  }
  if (threadIdx.x == 0) partial[blockIdx.x] = sm[0];
}

__global__ __launch_bounds__(256) void finalize_scales_kernel(
    const float* __restrict__ pgv, const float* __restrict__ pout,
    float* __restrict__ scl) {
  __shared__ float sm[256];
  float s = 0.f;
  for (int i = threadIdx.x; i < 2048; i += 256) s += pgv[i];
  sm[threadIdx.x] = s; __syncthreads();
  for (int o = 128; o > 0; o >>= 1) {
    if ((int)threadIdx.x < o) sm[threadIdx.x] += sm[threadIdx.x + o];
    __syncthreads();
  }
  const float sum_gv = sm[0];
  __syncthreads();
  s = 0.f;
  for (int i = threadIdx.x; i < 1024; i += 256) s += pout[i];
  sm[threadIdx.x] = s; __syncthreads();
  for (int o = 128; o > 0; o >>= 1) {
    if ((int)threadIdx.x < o) sm[threadIdx.x] += sm[threadIdx.x + o];
    __syncthreads();
  }
  const float sum_out = sm[0];
  if (threadIdx.x == 0) {
    const float mgv = fmaxf(sum_gv  / (float)((long long)NGV    * DMODEL), EPSF);
    const float mo  = fmaxf(sum_out / (float)((long long)DMODEL * HIDDEN), EPSF);
    scl[0] = mgv;  scl[1] = mo;  scl[2] = 1.f / mgv;  scl[3] = 1.f / mo;
  }
}

__global__ __launch_bounds__(256) void wquant_kernel(
    const float* __restrict__ w, int8_t* __restrict__ q, long long n,
    const float* __restrict__ scale_p) {
  const float s = *scale_p;
  long long i = (long long)blockIdx.x * blockDim.x + threadIdx.x;
  const long long stride = (long long)gridDim.x * blockDim.x;
  for (; i < n; i += stride)
    q[i] = (int8_t)(int)fminf(fmaxf(rintf(w[i] * s), -1.f), 1.f);
}

__global__ __launch_bounds__(256) void rmsnorm_actquant_kernel(
    const float* __restrict__ x, const float* __restrict__ w_norm,
    int8_t* __restrict__ qx, float* __restrict__ adq) {
  __shared__ float sm[256];
  __shared__ float h[DMODEL];
  const int t = blockIdx.x;
  const float* row = x + (long long)t * DMODEL;
  float ss = 0.f;
  for (int i = threadIdx.x; i < DMODEL; i += 256) { float v = row[i]; ss += v * v; }
  sm[threadIdx.x] = ss; __syncthreads();
  for (int o = 128; o > 0; o >>= 1) {
    if ((int)threadIdx.x < o) sm[threadIdx.x] += sm[threadIdx.x + o];
    __syncthreads();
  }
  const float rinv = rsqrtf(sm[0] / (float)DMODEL + EPSF);
  __syncthreads();
  float am = 0.f;
  for (int i = threadIdx.x; i < DMODEL; i += 256) {
    const float v = row[i] * rinv * w_norm[i];
    h[i] = v;
    am = fmaxf(am, fabsf(v));
  }
  sm[threadIdx.x] = am; __syncthreads();
  for (int o = 128; o > 0; o >>= 1) {
    if ((int)threadIdx.x < o) sm[threadIdx.x] = fmaxf(sm[threadIdx.x], sm[threadIdx.x + o]);
    __syncthreads();
  }
  const float amax  = fmaxf(sm[0], EPSF);
  const float scale = QBF / amax;
  for (int i = threadIdx.x; i < DMODEL; i += 256)
    qx[(long long)t * DMODEL + i] =
        (int8_t)(int)fminf(fmaxf(rintf(h[i] * scale), -128.f), 127.f);
  if (threadIdx.x == 0) adq[t] = amax / QBF;
}

union AFrag { uint64_t q[4]; v8i v; };
union BFrag { v4i x[2];     v8i v; };

__device__ __forceinline__ v8i load_a_frag(const int8_t* row, int k0, int hi) {
  AFrag a;
  const int off = k0 + hi * 16;
  a.q[0] = *(const uint64_t*)(row + off);
  a.q[1] = *(const uint64_t*)(row + off + 8);
  a.q[2] = *(const uint64_t*)(row + off + 32);
  a.q[3] = *(const uint64_t*)(row + off + 40);
  return a.v;
}
__device__ __forceinline__ v8i load_b_frag(const int8_t* row, int k0, int hi) {
  BFrag b;
  const int off = k0 + hi * 16;
  b.x[0] = *(const v4i*)(row + off);
  b.x[1] = *(const v4i*)(row + off + 32);
  return b.v;
}

__device__ __forceinline__ v8i wmma_iu8(v8i a, v8i b, v8i c) {
  return __builtin_amdgcn_wmma_i32_16x16x64_iu8(true, a, true, b, c, false, false);
}

__global__ __launch_bounds__(256) void gemm1_swiglu_kernel(
    const int8_t* __restrict__ qx, const int8_t* __restrict__ qw,
    const float* __restrict__ adq, const float* __restrict__ wdq_p,
    __hip_bfloat16* __restrict__ u) {
  const int lane = threadIdx.x & 31;
  const int wave = threadIdx.x >> 5;
  const int nl   = lane & 15;
  const int hi   = lane >> 4;
  const int mBase = blockIdx.x * 128 + (wave & 3) * 32;
  const int jBase = blockIdx.y * 64  + (wave >> 2) * 32;

  const int8_t* ar0 = qx + (long long)(mBase + nl) * DMODEL;
  const int8_t* ar1 = ar0 + 16 * DMODEL;
  const int8_t* bv0 = qw + (long long)(jBase + nl) * DMODEL;
  const int8_t* bv1 = bv0 + 16 * DMODEL;
  const int8_t* bg0 = qw + (long long)(HIDDEN + jBase + nl) * DMODEL;
  const int8_t* bg1 = bg0 + 16 * DMODEL;

  v8i cv00 = {0,0,0,0,0,0,0,0};
  v8i cv01 = cv00, cg00 = cv00, cg01 = cv00;
  v8i cv10 = cv00, cv11 = cv00, cg10 = cv00, cg11 = cv00;

  for (int k0 = 0; k0 < DMODEL; k0 += 64) {
    __builtin_prefetch(ar0 + k0 + 256, 0, 3);
    const v8i a0 = load_a_frag(ar0, k0, hi);
    const v8i a1 = load_a_frag(ar1, k0, hi);
    const v8i b0 = load_b_frag(bv0, k0, hi);
    const v8i b1 = load_b_frag(bv1, k0, hi);
    const v8i b2 = load_b_frag(bg0, k0, hi);
    const v8i b3 = load_b_frag(bg1, k0, hi);
    cv00 = wmma_iu8(a0, b0, cv00);
    cv01 = wmma_iu8(a0, b1, cv01);
    cg00 = wmma_iu8(a0, b2, cg00);
    cg01 = wmma_iu8(a0, b3, cg01);
    cv10 = wmma_iu8(a1, b0, cv10);
    cv11 = wmma_iu8(a1, b1, cv11);
    cg10 = wmma_iu8(a1, b2, cg10);
    cg11 = wmma_iu8(a1, b3, cg11);
  }

  const float wdq = *wdq_p;
#pragma unroll
  for (int r = 0; r < 8; ++r) {
    const int m0 = mBase + r + hi * 8;
    const int m1 = m0 + 16;
    const float f0 = adq[m0] * wdq;
    const float f1 = adq[m1] * wdq;
    const float v00 = (float)cv00[r] * f0, g00 = (float)cg00[r] * f0;
    const float v01 = (float)cv01[r] * f0, g01 = (float)cg01[r] * f0;
    const float v10 = (float)cv10[r] * f1, g10 = (float)cg10[r] * f1;
    const float v11 = (float)cv11[r] * f1, g11 = (float)cg11[r] * f1;
    const float u00 = v00 * (g00 / (1.f + __expf(-g00)));
    const float u01 = v01 * (g01 / (1.f + __expf(-g01)));
    const float u10 = v10 * (g10 / (1.f + __expf(-g10)));
    const float u11 = v11 * (g11 / (1.f + __expf(-g11)));
    u[(long long)m0 * HIDDEN + jBase + nl]      = __float2bfloat16(u00);
    u[(long long)m0 * HIDDEN + jBase + 16 + nl] = __float2bfloat16(u01);
    u[(long long)m1 * HIDDEN + jBase + nl]      = __float2bfloat16(u10);
    u[(long long)m1 * HIDDEN + jBase + 16 + nl] = __float2bfloat16(u11);
  }
}

__global__ __launch_bounds__(256) void uquant_kernel(
    const __hip_bfloat16* __restrict__ u, int8_t* __restrict__ qu,
    float* __restrict__ adq) {
  __shared__ float sm[256];
  const long long t = blockIdx.x;
  const __hip_bfloat16* row = u + t * HIDDEN;
  float am = 0.f;
  for (int i = threadIdx.x; i < HIDDEN; i += 256)
    am = fmaxf(am, fabsf(__bfloat162float(row[i])));
  sm[threadIdx.x] = am; __syncthreads();
  for (int o = 128; o > 0; o >>= 1) {
    if ((int)threadIdx.x < o) sm[threadIdx.x] = fmaxf(sm[threadIdx.x], sm[threadIdx.x + o]);
    __syncthreads();
  }
  const float amax  = fmaxf(sm[0], EPSF);
  const float scale = QBF / amax;
  for (int i = threadIdx.x; i < HIDDEN; i += 256) {
    const float v = __bfloat162float(row[i]);
    qu[t * HIDDEN + i] = (int8_t)(int)fminf(fmaxf(rintf(v * scale), -128.f), 127.f);
  }
  if (threadIdx.x == 0) adq[t] = amax / QBF;
}

__global__ __launch_bounds__(256) void gemm2_residual_kernel(
    const int8_t* __restrict__ qu, const int8_t* __restrict__ qw,
    const float* __restrict__ adq, const float* __restrict__ wdq_p,
    const float* __restrict__ x, float* __restrict__ out) {
  const int lane = threadIdx.x & 31;
  const int wave = threadIdx.x >> 5;
  const int nl   = lane & 15;
  const int hi   = lane >> 4;
  const int mBase = blockIdx.x * 128 + (wave & 3) * 32;
  const int nBase = blockIdx.y * 64  + (wave >> 2) * 32;

  const int8_t* ar0 = qu + (long long)(mBase + nl) * HIDDEN;
  const int8_t* ar1 = ar0 + 16 * HIDDEN;
  const int8_t* b0r = qw + (long long)(nBase + nl) * HIDDEN;
  const int8_t* b1r = b0r + 16 * HIDDEN;

  v8i c00 = {0,0,0,0,0,0,0,0};
  v8i c01 = c00, c10 = c00, c11 = c00;

  for (int k0 = 0; k0 < HIDDEN; k0 += 64) {
    __builtin_prefetch(ar0 + k0 + 256, 0, 3);
    const v8i a0 = load_a_frag(ar0, k0, hi);
    const v8i a1 = load_a_frag(ar1, k0, hi);
    const v8i b0 = load_b_frag(b0r, k0, hi);
    const v8i b1 = load_b_frag(b1r, k0, hi);
    c00 = wmma_iu8(a0, b0, c00);
    c01 = wmma_iu8(a0, b1, c01);
    c10 = wmma_iu8(a1, b0, c10);
    c11 = wmma_iu8(a1, b1, c11);
  }

  const float wdq = *wdq_p;
#pragma unroll
  for (int r = 0; r < 8; ++r) {
    const int m0 = mBase + r + hi * 8;
    const int m1 = m0 + 16;
    const float f0 = adq[m0] * wdq;
    const float f1 = adq[m1] * wdq;
    const long long o00 = (long long)m0 * DMODEL + nBase + nl;
    const long long o01 = o00 + 16;
    const long long o10 = (long long)m1 * DMODEL + nBase + nl;
    const long long o11 = o10 + 16;
    out[o00] = x[o00] + (float)c00[r] * f0;
    out[o01] = x[o01] + (float)c01[r] * f0;
    out[o10] = x[o10] + (float)c10[r] * f1;
    out[o11] = x[o11] + (float)c11[r] * f1;
  }
}

extern "C" void kernel_launch(void* const* d_in, const int* in_sizes, int n_in,
                              void* d_out, int out_size, void* d_ws, size_t ws_size,
                              hipStream_t stream) {
  const float* x      = (const float*)d_in[0];
  const float* w_norm = (const float*)d_in[1];
  const float* w_gv   = (const float*)d_in[2];
  const float* w_out  = (const float*)d_in[3];
  float* out = (float*)d_out;

  char* ws = (char*)d_ws;
  size_t off = 0;
  float* scl  = (float*)(ws + off); off += 256;
  float* pgv  = (float*)(ws + off); off += 2048 * sizeof(float);
  float* pout = (float*)(ws + off); off += 1024 * sizeof(float);
  off = (off + 255) & ~(size_t)255;
  int8_t* qw_gv  = (int8_t*)(ws + off); off += (size_t)NGV * DMODEL;
  int8_t* qw_out = (int8_t*)(ws + off); off += (size_t)DMODEL * HIDDEN;
  int8_t* qx     = (int8_t*)(ws + off); off += (size_t)TOKENS * DMODEL;
  float*  adq1   = (float*)(ws + off);  off += (size_t)TOKENS * sizeof(float);
  float*  adq2   = (float*)(ws + off);  off += (size_t)TOKENS * sizeof(float);
  __hip_bfloat16* u = (__hip_bfloat16*)(ws + off);
  off += (size_t)TOKENS * HIDDEN * sizeof(__hip_bfloat16);
  int8_t* qu     = (int8_t*)(ws + off); off += (size_t)TOKENS * HIDDEN;

  absum_partial_kernel<<<2048, 256, 0, stream>>>(w_gv,  (long long)NGV * DMODEL,    pgv);
  absum_partial_kernel<<<1024, 256, 0, stream>>>(w_out, (long long)DMODEL * HIDDEN, pout);
  finalize_scales_kernel<<<1, 256, 0, stream>>>(pgv, pout, scl);

  wquant_kernel<<<4096, 256, 0, stream>>>(w_gv,  qw_gv,  (long long)NGV * DMODEL,    scl + 2);
  wquant_kernel<<<2048, 256, 0, stream>>>(w_out, qw_out, (long long)DMODEL * HIDDEN, scl + 3);

  rmsnorm_actquant_kernel<<<TOKENS, 256, 0, stream>>>(x, w_norm, qx, adq1);

  gemm1_swiglu_kernel<<<dim3(TOKENS / 128, HIDDEN / 64), 256, 0, stream>>>(
      qx, qw_gv, adq1, scl + 0, u);

  uquant_kernel<<<TOKENS, 256, 0, stream>>>(u, qu, adq2);

  gemm2_residual_kernel<<<dim3(TOKENS / 128, DMODEL / 64), 256, 0, stream>>>(
      qu, qw_out, adq2, scl + 1, x, out);
}
